// UGCALayer_32220844655400
// MI455X (gfx1250) — matrix operand order repaired and checked by reading, not compiled, not run on hardware
//
#include <hip/hip_runtime.h>
#include <hip/hip_bf16.h>
#include <math.h>

typedef __bf16 bf16_t;
typedef __attribute__((ext_vector_type(16))) __bf16 v16bf;
typedef __attribute__((ext_vector_type(8)))  __bf16 bf16x8;
typedef __attribute__((ext_vector_type(4)))  __bf16 bf16x4;
typedef __attribute__((ext_vector_type(8)))  float   v8f;

#define DMODEL 512
#define NHEADS 8
#define HDIM   64

__device__ __forceinline__ v8f vzero8() {
    v8f z = { 0.f,0.f,0.f,0.f,0.f,0.f,0.f,0.f };
    return z;
}

__device__ __forceinline__ v8f wmma_bf16(v16bf a, v16bf b, v8f c) {
    return __builtin_amdgcn_wmma_f32_16x16x32_bf16(false, a, false, b, (short)0, c, false, false);
}

// A-fragment (16x32, MxK): lane&15 = row, halves at k = hi*8..+7 and 16+hi*8..+7
__device__ __forceinline__ v16bf load_a_frag(const bf16_t* __restrict__ base, int ld) {
    const int lane = threadIdx.x & 31;
    const bf16_t* p = base + (size_t)(lane & 15) * (size_t)ld + (size_t)((lane >> 4) << 3);
    union { v16bf v; bf16x8 h[2]; } u;
    u.h[0] = *(const bf16x8*)(p);
    u.h[1] = *(const bf16x8*)(p + 16);
    return u.v;
}

// B-fragment (32x16, KxN) for NT GEMM: lane&15 = column (= row of row-major W),
// halves at k = hi*8..+7 and 16+hi*8..+7
__device__ __forceinline__ v16bf load_b_frag(const bf16_t* __restrict__ base, int ld) {
    const int lane = threadIdx.x & 31;
    const bf16_t* p = base + (size_t)(lane & 15) * (size_t)ld + (size_t)((lane >> 4) << 3);
    union { v16bf v; bf16x8 h[2]; } u;
    u.h[0] = *(const bf16x8*)(p);
    u.h[1] = *(const bf16x8*)(p + 16);
    return u.v;
}

__device__ __forceinline__ float gelu_exact(float x) {
    return 0.5f * x * (1.0f + erff(x * 0.70710678118654752f));
}

__device__ __forceinline__ float softplusf(float x) {
    return (x > 20.f) ? x : log1pf(__expf(x));
}

// ---------------------------------------------------------------------------
// fp32 -> bf16 elementwise conversion (vectorized x4)
// ---------------------------------------------------------------------------
__global__ void cvt_f32_bf16(const float* __restrict__ s, bf16_t* __restrict__ d, int n4) {
    int i = blockIdx.x * blockDim.x + threadIdx.x;
    if (i >= n4) return;
    float4 v = ((const float4*)s)[i];
    bf16x4 o = { (bf16_t)v.x, (bf16_t)v.y, (bf16_t)v.z, (bf16_t)v.w };
    ((bf16x4*)d)[i] = o;
}

// ---------------------------------------------------------------------------
// Uncertainty gate: one wave per token, 4 dot products of length 512,
// evidential NIG gate -> log(clip(g,1e-3,1)+1e-9)
// ---------------------------------------------------------------------------
__global__ __launch_bounds__(256) void gate_kernel(
    const float* __restrict__ X, const float* __restrict__ gw, const float* __restrict__ gb,
    float* __restrict__ lgout, int tokens)
{
    const int lane = threadIdx.x & 31;
    const int t = blockIdx.x * 8 + (threadIdx.x >> 5);
    if (t >= tokens) return;
    const float* x = X + (size_t)t * DMODEL;
    float a0 = 0.f, a1 = 0.f, a2 = 0.f, a3 = 0.f;
    for (int i = lane; i < DMODEL; i += 32) {
        float xv = x[i];
        a0 += xv * gw[i];
        a1 += xv * gw[DMODEL + i];
        a2 += xv * gw[2 * DMODEL + i];
        a3 += xv * gw[3 * DMODEL + i];
    }
    #pragma unroll
    for (int o = 1; o < 32; o <<= 1) {
        a0 += __shfl_xor(a0, o, 32);
        a1 += __shfl_xor(a1, o, 32);
        a2 += __shfl_xor(a2, o, 32);
        a3 += __shfl_xor(a3, o, 32);
    }
    if (lane == 0) {
        float mu = a0 + gb[0];
        float vv = softplusf(a1 + gb[1]) + 1e-6f;
        float al = softplusf(a2 + gb[2]) + 1.0f + 1e-6f;
        float be = softplusf(a3 + gb[3]) + 1e-6f;
        float s2 = be / (vv * (al - 1.0f));
        float g = (1.0f / (1.0f + __expf(-mu))) * __expf(-s2);
        g = fminf(fmaxf(g, 1e-3f), 1.0f);
        lgout[t] = __logf(g + 1e-9f);
    }
}

// ---------------------------------------------------------------------------
// NT GEMM:  C[M,N] = A[M,K](bf16,rm) * W[N,K](bf16,rm)^T + bias
// block = 256 thr = 8 waves; block tile 64(M) x 256(N); wave tile 32x64
// act: 0=none, 1=exact gelu.  Cf (fp32) / Cb (bf16) nullable.
// M%64==0, N%256==0, K%32==0 for all call sites.
// ---------------------------------------------------------------------------
__global__ __launch_bounds__(256) void gemm_nt_wmma(
    const bf16_t* __restrict__ A, const bf16_t* __restrict__ W,
    const float* __restrict__ bias,
    float* __restrict__ Cf, bf16_t* __restrict__ Cb,
    int M, int N, int K, int act)
{
    const int lane = threadIdx.x & 31;
    const int w    = threadIdx.x >> 5;
    const int m0 = blockIdx.y * 64 + (w >> 2) * 32;
    const int n0 = blockIdx.x * 256 + (w & 3) * 64;

    v8f acc[2][4];
    #pragma unroll
    for (int i = 0; i < 2; ++i)
        #pragma unroll
        for (int j = 0; j < 4; ++j) acc[i][j] = vzero8();

    const bf16_t* Abase0 = A + (size_t)m0 * (size_t)K;
    const bf16_t* Abase1 = A + (size_t)(m0 + 16) * (size_t)K;

    for (int k0 = 0; k0 < K; k0 += 32) {
        v16bf a0 = load_a_frag(Abase0 + k0, K);
        v16bf a1 = load_a_frag(Abase1 + k0, K);
        #pragma unroll
        for (int j = 0; j < 4; ++j) {
            v16bf bj = load_b_frag(W + (size_t)(n0 + j * 16) * (size_t)K + k0, K);
            acc[0][j] = wmma_bf16(a0, bj, acc[0][j]);
            acc[1][j] = wmma_bf16(a1, bj, acc[1][j]);
        }
    }

    const int col = lane & 15, hi = lane >> 4;
    #pragma unroll
    for (int j = 0; j < 4; ++j) {
        const int n = n0 + j * 16 + col;
        const float bv = bias ? bias[n] : 0.f;
        #pragma unroll
        for (int i = 0; i < 2; ++i) {
            #pragma unroll
            for (int r = 0; r < 8; ++r) {
                const int row = m0 + i * 16 + hi * 8 + r;
                float v = acc[i][j][r] + bv;
                if (act == 1) v = gelu_exact(v);
                const size_t off = (size_t)row * (size_t)N + n;
                if (Cf) Cf[off] = v;
                if (Cb) Cb[off] = (bf16_t)v;
            }
        }
    }
}

// ---------------------------------------------------------------------------
// V transpose: Vt[(b*8+h)*64+d][k] = V[(b*Lk+k)*512 + h*64 + d]
// ---------------------------------------------------------------------------
__global__ void vtrans_kernel(const bf16_t* __restrict__ V, bf16_t* __restrict__ Vt,
                              int Bsz, int Lk) {
    size_t idx = (size_t)blockIdx.x * blockDim.x + threadIdx.x;
    size_t total = (size_t)Bsz * DMODEL * (size_t)Lk;
    if (idx >= total) return;
    int k = (int)(idx % Lk);
    size_t t = idx / Lk;
    int d = (int)(t % HDIM); t /= HDIM;
    int h = (int)(t % NHEADS);
    int b = (int)(t / NHEADS);
    Vt[idx] = V[((size_t)b * Lk + k) * DMODEL + h * HDIM + d];
}

// ---------------------------------------------------------------------------
// Flash cross-attention, one wave per (b, h, 16-query tile).
// Q,K: [tokens, 512] bf16 (head h at column h*64); Vt: [B,H,64,Lk] bf16.
// scores = QK^T/8 + lg[key]  (or -1e9 where mask==0), streaming softmax,
// out = P V / l, written bf16 to Out [tokens, 512].
// ---------------------------------------------------------------------------
__global__ __launch_bounds__(256) void flash_attn_wmma(
    const bf16_t* __restrict__ Q, const bf16_t* __restrict__ Kmat,
    const bf16_t* __restrict__ Vt, const float* __restrict__ lg,
    const int* __restrict__ mask, bf16_t* __restrict__ Out,
    int Lq, int Lk)
{
    __shared__ alignas(16) bf16_t pbuf[8][16 * 32];
    const int lane = threadIdx.x & 31;
    const int w    = threadIdx.x >> 5;
    const int nq = Lq >> 4;
    const int wid = blockIdx.x * 8 + w;
    const int qt = wid % nq;
    const int h  = (wid / nq) & (NHEADS - 1);
    const int b  = wid / (nq * NHEADS);
    const int col = lane & 15, hi = lane >> 4;

    const bf16_t* qbase = Q + ((size_t)(b * Lq + qt * 16)) * DMODEL + h * HDIM;
    v16bf qa0 = load_a_frag(qbase, DMODEL);        // head dims 0..31
    v16bf qa1 = load_a_frag(qbase + 32, DMODEL);   // head dims 32..63

    v8f o[4];
    #pragma unroll
    for (int c = 0; c < 4; ++c) o[c] = vzero8();
    float mrun[8], lrun[8];
    #pragma unroll
    for (int r = 0; r < 8; ++r) { mrun[r] = -1e30f; lrun[r] = 0.f; }

    const bf16_t* kbase = Kmat + (size_t)(b * Lk) * DMODEL + h * HDIM;
    const bf16_t* vtb   = Vt + ((size_t)(b * NHEADS + h)) * HDIM * (size_t)Lk;
    const float* lgb = lg + b * Lk;
    const int*   mb  = mask + b * Lk;
    bf16_t* pl = &pbuf[w][0];

    for (int kb = 0; kb < Lk; kb += 32) {
        v8f s0 = vzero8(), s1 = vzero8();
        {
            v16bf k00 = load_b_frag(kbase + (size_t)kb * DMODEL, DMODEL);
            v16bf k01 = load_b_frag(kbase + (size_t)kb * DMODEL + 32, DMODEL);
            s0 = wmma_bf16(qa0, k00, s0);
            s0 = wmma_bf16(qa1, k01, s0);
            v16bf k10 = load_b_frag(kbase + (size_t)(kb + 16) * DMODEL, DMODEL);
            v16bf k11 = load_b_frag(kbase + (size_t)(kb + 16) * DMODEL + 32, DMODEL);
            s1 = wmma_bf16(qa0, k10, s1);
            s1 = wmma_bf16(qa1, k11, s1);
        }
        const int k0i = kb + col, k1i = kb + 16 + col;
        const float lg0 = lgb[k0i], lg1 = lgb[k1i];
        const bool ok0 = (mb[k0i] != 0), ok1 = (mb[k1i] != 0);
        #pragma unroll
        for (int r = 0; r < 8; ++r) {
            float v0 = ok0 ? (s0[r] * 0.125f + lg0) : -1e9f;
            float v1 = ok1 ? (s1[r] * 0.125f + lg1) : -1e9f;
            float t = fmaxf(v0, v1);
            t = fmaxf(t, __shfl_xor(t, 1, 32));
            t = fmaxf(t, __shfl_xor(t, 2, 32));
            t = fmaxf(t, __shfl_xor(t, 4, 32));
            t = fmaxf(t, __shfl_xor(t, 8, 32));
            const float mnew = fmaxf(mrun[r], t);
            const float sc = __expf(mrun[r] - mnew);
            mrun[r] = mnew;
            const float e0 = __expf(v0 - mnew);
            const float e1 = __expf(v1 - mnew);
            float rs = e0 + e1;
            rs += __shfl_xor(rs, 1, 32);
            rs += __shfl_xor(rs, 2, 32);
            rs += __shfl_xor(rs, 4, 32);
            rs += __shfl_xor(rs, 8, 32);
            lrun[r] = lrun[r] * sc + rs;
            o[0][r] = o[0][r] * sc;
            o[1][r] = o[1][r] * sc;
            o[2][r] = o[2][r] * sc;
            o[3][r] = o[3][r] * sc;
            const int prow = hi * 8 + r;                 // query row in tile
            pl[prow * 32 + col]      = (bf16_t)e0;       // key kb+col
            pl[prow * 32 + 16 + col] = (bf16_t)e1;       // key kb+16+col
        }
        // reload P as A-fragment (wave-private LDS; DS ops are in-order per wave)
        union { v16bf v; bf16x8 hh[2]; } pu;
        pu.hh[0] = *(const bf16x8*)&pl[col * 32 + hi * 8];
        pu.hh[1] = *(const bf16x8*)&pl[col * 32 + 16 + hi * 8];
        #pragma unroll
        for (int c = 0; c < 4; ++c) {
            v16bf vf = load_b_frag(vtb + (size_t)(c * 16) * (size_t)Lk + kb, Lk);
            o[c] = wmma_bf16(pu.v, vf, o[c]);
        }
    }

    #pragma unroll
    for (int c = 0; c < 4; ++c) {
        #pragma unroll
        for (int r = 0; r < 8; ++r) {
            const int qrow = qt * 16 + hi * 8 + r;
            const int d = c * 16 + col;
            Out[((size_t)(b * Lq + qrow)) * DMODEL + h * HDIM + d] =
                (bf16_t)(o[c][r] / lrun[r]);
        }
    }
}

// ---------------------------------------------------------------------------
// y = LN(resid + x) * gamma + beta ; one wave per 512-wide row
// ---------------------------------------------------------------------------
__global__ __launch_bounds__(256) void add_ln_kernel(
    const float* __restrict__ resid, const float* __restrict__ x,
    const float* __restrict__ gamma, const float* __restrict__ beta,
    float* __restrict__ outf, bf16_t* __restrict__ outb, int rows)
{
    const int lane = threadIdx.x & 31;
    const int row = blockIdx.x * 8 + (threadIdx.x >> 5);
    if (row >= rows) return;
    const float4* r4 = (const float4*)(resid + (size_t)row * DMODEL);
    const float4* x4 = (const float4*)(x + (size_t)row * DMODEL);
    float v[16];
    float s = 0.f;
    #pragma unroll
    for (int i = 0; i < 4; ++i) {
        float4 a = r4[lane + i * 32];
        float4 bq = x4[lane + i * 32];
        v[i * 4 + 0] = a.x + bq.x; v[i * 4 + 1] = a.y + bq.y;
        v[i * 4 + 2] = a.z + bq.z; v[i * 4 + 3] = a.w + bq.w;
        s += v[i * 4 + 0] + v[i * 4 + 1] + v[i * 4 + 2] + v[i * 4 + 3];
    }
    s += __shfl_xor(s, 1, 32); s += __shfl_xor(s, 2, 32); s += __shfl_xor(s, 4, 32);
    s += __shfl_xor(s, 8, 32); s += __shfl_xor(s, 16, 32);
    const float mean = s * (1.f / 512.f);
    float vs = 0.f;
    #pragma unroll
    for (int i = 0; i < 16; ++i) { float d = v[i] - mean; vs += d * d; }
    vs += __shfl_xor(vs, 1, 32); vs += __shfl_xor(vs, 2, 32); vs += __shfl_xor(vs, 4, 32);
    vs += __shfl_xor(vs, 8, 32); vs += __shfl_xor(vs, 16, 32);
    const float rstd = rsqrtf(vs * (1.f / 512.f) + 1e-5f);
    #pragma unroll
    for (int i = 0; i < 4; ++i) {
        const int c0 = (lane + i * 32) * 4;
        float o0 = (v[i * 4 + 0] - mean) * rstd * gamma[c0 + 0] + beta[c0 + 0];
        float o1 = (v[i * 4 + 1] - mean) * rstd * gamma[c0 + 1] + beta[c0 + 1];
        float o2 = (v[i * 4 + 2] - mean) * rstd * gamma[c0 + 2] + beta[c0 + 2];
        float o3 = (v[i * 4 + 3] - mean) * rstd * gamma[c0 + 3] + beta[c0 + 3];
        if (outf) {
            float4 ov = { o0, o1, o2, o3 };
            ((float4*)(outf + (size_t)row * DMODEL))[lane + i * 32] = ov;
        }
        if (outb) {
            bf16x4 ob = { (bf16_t)o0, (bf16_t)o1, (bf16_t)o2, (bf16_t)o3 };
            ((bf16x4*)(outb + (size_t)row * DMODEL))[lane + i * 32] = ob;
        }
    }
}

// ---------------------------------------------------------------------------
// Host orchestration
// ---------------------------------------------------------------------------
extern "C" void kernel_launch(void* const* d_in, const int* in_sizes, int n_in,
                              void* d_out, int out_size, void* d_ws, size_t ws_size,
                              hipStream_t stream)
{
    (void)in_sizes; (void)n_in; (void)out_size; (void)ws_size;
    const int B = 16, Ld = 256, Lp = 1024, D = 512, F = 2048;
    const int Md = B * Ld;   // 4096 drug tokens
    const int Mp = B * Lp;   // 16384 protein tokens

    const float* h_d = (const float*)d_in[0];
    const float* h_p = (const float*)d_in[1];
    const int* mask_d = (const int*)d_in[38];
    const int* mask_p = (const int*)d_in[39];
    float* out_hd = (float*)d_out;
    float* out_hp = (float*)d_out + (size_t)Md * D;

    // ---- workspace carving ----
    char* wp = (char*)d_ws;
    auto carve = [&](size_t bytes) -> void* {
        void* r = (void*)wp;
        wp += (bytes + 255) & ~(size_t)255;
        return r;
    };
    const size_t szDD = (size_t)D * D, szFD = (size_t)F * D;

    bf16_t* hd_bf  = (bf16_t*)carve((size_t)Md * D * 2);
    bf16_t* hp_bf  = (bf16_t*)carve((size_t)Mp * D * 2);
    bf16_t* w_qd  = (bf16_t*)carve(szDD * 2);
    bf16_t* w_kp  = (bf16_t*)carve(szDD * 2);
    bf16_t* w_vp  = (bf16_t*)carve(szDD * 2);
    bf16_t* w_qp  = (bf16_t*)carve(szDD * 2);
    bf16_t* w_kd  = (bf16_t*)carve(szDD * 2);
    bf16_t* w_vd  = (bf16_t*)carve(szDD * 2);
    bf16_t* w_od  = (bf16_t*)carve(szDD * 2);
    bf16_t* w_op  = (bf16_t*)carve(szDD * 2);
    bf16_t* w_fd1 = (bf16_t*)carve(szFD * 2);
    bf16_t* w_fd2 = (bf16_t*)carve(szFD * 2);
    bf16_t* w_fp1 = (bf16_t*)carve(szFD * 2);
    bf16_t* w_fp2 = (bf16_t*)carve(szFD * 2);
    bf16_t* Qd    = (bf16_t*)carve((size_t)Md * D * 2);
    bf16_t* Kp    = (bf16_t*)carve((size_t)Mp * D * 2);
    bf16_t* Vp    = (bf16_t*)carve((size_t)Mp * D * 2);
    bf16_t* Vpt   = (bf16_t*)carve((size_t)Mp * D * 2);
    bf16_t* attnd = (bf16_t*)carve((size_t)Md * D * 2);
    bf16_t* Qp    = (bf16_t*)carve((size_t)Mp * D * 2);
    bf16_t* Kd    = (bf16_t*)carve((size_t)Md * D * 2);
    bf16_t* Vd    = (bf16_t*)carve((size_t)Md * D * 2);
    bf16_t* Vdt   = (bf16_t*)carve((size_t)Md * D * 2);
    bf16_t* attnp = (bf16_t*)carve((size_t)Mp * D * 2);
    bf16_t* hd1_bf = (bf16_t*)carve((size_t)Md * D * 2);
    bf16_t* hp1_bf = (bf16_t*)carve((size_t)Mp * D * 2);
    bf16_t* hid    = (bf16_t*)carve((size_t)Mp * F * 2);
    float* lg_d  = (float*)carve((size_t)Md * 4);
    float* lg_p  = (float*)carve((size_t)Mp * 4);
    float* hd1_f = (float*)carve((size_t)Md * D * 4);
    float* hp1_f = (float*)carve((size_t)Mp * D * 4);
    float* projf = (float*)carve((size_t)Mp * D * 4);

    // ---- helpers ----
    auto cvt = [&](const void* src, bf16_t* dst, size_t n) {
        int n4 = (int)(n / 4);
        cvt_f32_bf16<<<dim3((n4 + 255) / 256), dim3(256), 0, stream>>>((const float*)src, dst, n4);
    };
    auto gemm = [&](const bf16_t* A, const bf16_t* W, const float* bias,
                    float* Cf, bf16_t* Cb, int M, int N, int K, int act) {
        gemm_nt_wmma<<<dim3(N / 256, M / 64), dim3(256), 0, stream>>>(A, W, bias, Cf, Cb, M, N, K, act);
    };
    auto add_ln = [&](const float* resid, const float* x, int gi, int bi,
                      float* of, bf16_t* ob, int rows) {
        add_ln_kernel<<<dim3(rows / 8), dim3(256), 0, stream>>>(
            resid, x, (const float*)d_in[gi], (const float*)d_in[bi], of, ob, rows);
    };

    // ---- 1) bf16 conversions ----
    cvt(d_in[0], hd_bf, (size_t)Md * D);
    cvt(d_in[1], hp_bf, (size_t)Mp * D);
    cvt(d_in[6],  w_qd,  szDD);  cvt(d_in[8],  w_kp,  szDD);
    cvt(d_in[10], w_vp,  szDD);  cvt(d_in[12], w_qp,  szDD);
    cvt(d_in[14], w_kd,  szDD);  cvt(d_in[16], w_vd,  szDD);
    cvt(d_in[18], w_od,  szDD);  cvt(d_in[20], w_op,  szDD);
    cvt(d_in[22], w_fd1, szFD);  cvt(d_in[24], w_fd2, szFD);
    cvt(d_in[26], w_fp1, szFD);  cvt(d_in[28], w_fp2, szFD);

    // ---- 2) gates (from ORIGINAL inputs) ----
    gate_kernel<<<dim3(Md / 8), dim3(256), 0, stream>>>(
        h_d, (const float*)d_in[2], (const float*)d_in[3], lg_d, Md);
    gate_kernel<<<dim3(Mp / 8), dim3(256), 0, stream>>>(
        h_p, (const float*)d_in[4], (const float*)d_in[5], lg_p, Mp);

    // ---- 3) drug -> protein attention ----
    gemm(hd_bf, w_qd, (const float*)d_in[7],  nullptr, Qd, Md, D, D, 0);
    gemm(hp_bf, w_kp, (const float*)d_in[9],  nullptr, Kp, Mp, D, D, 0);
    gemm(hp_bf, w_vp, (const float*)d_in[11], nullptr, Vp, Mp, D, D, 0);
    {
        size_t tot = (size_t)B * D * Lp;
        vtrans_kernel<<<dim3((unsigned)(tot / 256)), dim3(256), 0, stream>>>(Vp, Vpt, B, Lp);
    }
    flash_attn_wmma<<<dim3(B * NHEADS * (Ld / 16) / 8), dim3(256), 0, stream>>>(
        Qd, Kp, Vpt, lg_p, mask_p, attnd, Ld, Lp);
    gemm(attnd, w_od, (const float*)d_in[19], projf, nullptr, Md, D, D, 0);
    add_ln(h_d, projf, 30, 31, hd1_f, hd1_bf, Md);   // nd1

    // ---- 4) protein -> (updated) drug attention ----
    gemm(hp_bf,  w_qp, (const float*)d_in[13], nullptr, Qp, Mp, D, D, 0);
    gemm(hd1_bf, w_kd, (const float*)d_in[15], nullptr, Kd, Md, D, D, 0);
    gemm(hd1_bf, w_vd, (const float*)d_in[17], nullptr, Vd, Md, D, D, 0);
    {
        size_t tot = (size_t)B * D * Ld;
        vtrans_kernel<<<dim3((unsigned)(tot / 256)), dim3(256), 0, stream>>>(Vd, Vdt, B, Ld);
    }
    flash_attn_wmma<<<dim3(B * NHEADS * (Lp / 16) / 8), dim3(256), 0, stream>>>(
        Qp, Kd, Vdt, lg_d, mask_d, attnp, Lp, Ld);
    gemm(attnp, w_op, (const float*)d_in[21], projf, nullptr, Mp, D, D, 0);
    add_ln(h_p, projf, 34, 35, hp1_f, hp1_bf, Mp);   // np1

    // ---- 5) FFN drug -> final h_d ----
    gemm(hd1_bf, w_fd1, (const float*)d_in[23], nullptr, hid, Md, F, D, 1);
    gemm(hid,    w_fd2, (const float*)d_in[25], projf,  nullptr, Md, D, F, 0);
    add_ln(hd1_f, projf, 32, 33, out_hd, nullptr, Md);  // nd2 -> d_out

    // ---- 6) FFN protein -> final h_p ----
    gemm(hp1_bf, w_fp1, (const float*)d_in[27], nullptr, hid, Mp, F, D, 1);
    gemm(hid,    w_fp2, (const float*)d_in[29], projf,  nullptr, Mp, D, F, 0);
    add_ln(hp1_f, projf, 36, 37, out_hp, nullptr, Mp);  // np2 -> d_out
}
